// HLSA_18339510354360
// MI455X (gfx1250) — matrix operand order repaired and checked by reading, not compiled, not run on hardware
//
#include <hip/hip_runtime.h>
#include <hip/hip_bf16.h>

// ---------------------------------------------------------------------------
// Problem constants (fixed by the reference)
// ---------------------------------------------------------------------------
#define BB      8
#define NN      4096
#define CC      512
#define HEADS   8
#define DH      64          // head dim
#define NS      256         // (64/4)*(64/4) reduced tokens
#define KCONV   8192        // C * SR * SR
#define SCALE   0.044194173824159216f   // heads^-0.5 * d^-0.5

typedef __attribute__((ext_vector_type(16))) __bf16 v16bf;
typedef __attribute__((ext_vector_type(8)))  float  v8f;

// CDNA5 async global->LDS staging (ASYNCcnt-tracked), guarded so the file
// still compiles if this toolchain lacks the builtins.
#if defined(__AMDGCN__) && \
    __has_builtin(__builtin_amdgcn_global_load_async_to_lds_b32) && \
    __has_builtin(__builtin_amdgcn_s_wait_asynccnt)
#define HAVE_ASYNC_LDS 1
#define AS1 __attribute__((address_space(1)))
#define AS3 __attribute__((address_space(3)))
#endif

static __device__ inline __bf16 f2bf(float f) {
    union { float f; unsigned u; } a; a.f = f;
    unsigned r = a.u + 0x7FFFu + ((a.u >> 16) & 1u);   // round-to-nearest-even
    union { unsigned short s; __bf16 b; } o; o.s = (unsigned short)(r >> 16);
    return o.b;
}

// ---------------------------------------------------------------------------
// Packing kernels (fp32 -> bf16, with layout rearrangement where needed)
// ---------------------------------------------------------------------------
__global__ void pack_bf16(const float* __restrict__ s, __bf16* __restrict__ d, int n) {
    int i = blockIdx.x * 256 + threadIdx.x;
    if (i < n) d[i] = f2bf(s[i]);
}

// conv_w [O=512, I=512, kh=4, kw=4] (OIHW) -> B-matrix [K=8192, O=512],
// K index = (kh*4+kw)*512 + i  (matches im2row ordering below)
__global__ void pack_convw(const float* __restrict__ w, __bf16* __restrict__ d) {
    int idx = blockIdx.x * 256 + threadIdx.x;       // k*512 + o
    if (idx >= KCONV * CC) return;
    int k = idx >> 9, o = idx & 511;
    int i = k & 511, khkw = k >> 9;
    int kh = khkw >> 2, kw = khkw & 3;
    d[idx] = f2bf(w[((size_t)(o * 512 + i) * 4 + kh) * 4 + kw]);
}

// x2 [B, N=HW, C] -> im2row A [B*Ns, 8192] with k = (kh*4+kw)*512 + i
__global__ void pack_patches(const float* __restrict__ x2, __bf16* __restrict__ d) {
    int idx = blockIdx.x * 256 + threadIdx.x;       // r*8192 + k
    if (idx >= BB * NS * KCONV) return;
    int r = idx >> 13, k = idx & 8191;
    int b = r >> 8, pp = r & 255;
    int ph = pp >> 4, pw = pp & 15;
    int i = k & 511, khkw = k >> 9;
    int kh = khkw >> 2, kw = khkw & 3;
    int n = (ph * 4 + kh) * 64 + (pw * 4 + kw);
    d[idx] = f2bf(x2[((size_t)b * NN + n) * CC + i]);
}

// ---------------------------------------------------------------------------
// Generic bf16 WMMA GEMM:  C[M,N] = concat_K(A0,A1)[M,K] * B[K,N] + bias
//  - 128x128x32 tiles, 256 threads (8 waves), each wave = 64x32 of output
//  - Both tiles staged via CDNA5 async global->LDS b32 copies (ASYNCcnt),
//    A row-major, B row-major [32][132]; fragment gathers follow the
//    documented wave32 WMMA VGPR layouts.
// ---------------------------------------------------------------------------
template <bool OUT_BF16>
__global__ __launch_bounds__(256)
void gemm_bf16(const __bf16* __restrict__ A0, const __bf16* __restrict__ A1,
               int splitK, int lda0, int lda1,
               const __bf16* __restrict__ Bm, const float* __restrict__ bias,
               void* __restrict__ Cout, int M, int K, int Nn)
{
    __shared__ __bf16 sA[128][40];    // 128 rows x 32 K (+pad)
    __shared__ __bf16 sB[32][132];    // 32 K x 128 cols (+pad)

    const int tid  = threadIdx.x;
    const int lane = tid & 31, wave = tid >> 5;
    const int waveM = wave >> 2, waveN = wave & 3;   // 2 x 4 wave grid
    const int m0 = blockIdx.y * 128, n0 = blockIdx.x * 128;
    const int r = lane & 15, half = lane >> 4;

    v8f acc[4][2];
#pragma unroll
    for (int mi = 0; mi < 4; ++mi)
#pragma unroll
        for (int ni = 0; ni < 2; ++ni)
#pragma unroll
            for (int j = 0; j < 8; ++j) acc[mi][ni][j] = 0.0f;

    for (int k0 = 0; k0 < K; k0 += 32) {
        __syncthreads();
        // --- A tile: 128x32 bf16 = 2048 dwords, 8 per thread -------------
#pragma unroll
        for (int it = 0; it < 8; ++it) {
            int e = tid + it * 256;
            int rr = e >> 4, c2 = (e & 15) * 2;
            int gk = k0 + c2;
            const __bf16* gp = (gk < splitK)
                ? A0 + (size_t)(m0 + rr) * lda0 + gk
                : A1 + (size_t)(m0 + rr) * lda1 + (gk - splitK);
#ifdef HAVE_ASYNC_LDS
            __builtin_amdgcn_global_load_async_to_lds_b32(
                (AS1 int*)(void*)gp, (AS3 int*)(void*)&sA[rr][c2], 0, 0);
#else
            *(unsigned*)&sA[rr][c2] = *(const unsigned*)gp;
#endif
        }
        // --- B tile: 32x128 bf16 = 2048 dwords, row-major ---------------
#pragma unroll
        for (int it = 0; it < 8; ++it) {
            int e = tid + it * 256;
            int kk = e >> 6, c2 = (e & 63) * 2;
            const __bf16* gp = Bm + (size_t)(k0 + kk) * Nn + n0 + c2;
#ifdef HAVE_ASYNC_LDS
            __builtin_amdgcn_global_load_async_to_lds_b32(
                (AS1 int*)(void*)gp, (AS3 int*)(void*)&sB[kk][c2], 0, 0);
#else
            *(unsigned*)&sB[kk][c2] = *(const unsigned*)gp;
#endif
        }
#ifdef HAVE_ASYNC_LDS
        __builtin_amdgcn_s_wait_asynccnt(0);
#endif
        __syncthreads();

        // --- fragments (documented 16-bit A/B layouts, wave32) -----------
        v16bf afr[4], bfr[2];
#pragma unroll
        for (int mi = 0; mi < 4; ++mi) {
            int row = waveM * 64 + mi * 16 + r;
#pragma unroll
            for (int j = 0; j < 8; ++j) {
                int kk = ((j >= 4) ? 16 : 0) + half * 8 + (j & 3) * 2;
                afr[mi][2 * j]     = sA[row][kk];
                afr[mi][2 * j + 1] = sA[row][kk + 1];
            }
        }
#pragma unroll
        for (int ni = 0; ni < 2; ++ni) {
            int col = waveN * 32 + ni * 16 + r;
#pragma unroll
            for (int j = 0; j < 8; ++j) {
                int kk = ((j >= 4) ? 16 : 0) + half * 8 + (j & 3) * 2;
                bfr[ni][2 * j]     = sB[kk][col];
                bfr[ni][2 * j + 1] = sB[kk + 1][col];
            }
        }
#pragma unroll
        for (int mi = 0; mi < 4; ++mi)
#pragma unroll
            for (int ni = 0; ni < 2; ++ni)
                acc[mi][ni] = __builtin_amdgcn_wmma_f32_16x16x32_bf16(
                    false, afr[mi], false, bfr[ni], (short)0, acc[mi][ni],
                    false, false);
    }

    // --- epilogue: C layout row = half*8 + j, col = lane%16 --------------
#pragma unroll
    for (int mi = 0; mi < 4; ++mi)
#pragma unroll
        for (int ni = 0; ni < 2; ++ni) {
            int col = n0 + waveN * 32 + ni * 16 + r;
            float bv = bias ? bias[col] : 0.0f;
#pragma unroll
            for (int j = 0; j < 8; ++j) {
                int row = m0 + waveM * 64 + mi * 16 + half * 8 + j;
                float v = acc[mi][ni][j] + bv;
                if (OUT_BF16)
                    ((__bf16*)Cout)[(size_t)row * Nn + col] = f2bf(v);
                else
                    ((float*)Cout)[(size_t)row * Nn + col] = v;
            }
        }
}

// ---------------------------------------------------------------------------
// LayerNorm over C=512 per reduced token, fp32 in -> bf16 out (gamma/beta)
// ---------------------------------------------------------------------------
__global__ __launch_bounds__(256)
void ln_kernel(const float* __restrict__ x, const float* __restrict__ g,
               const float* __restrict__ b, __bf16* __restrict__ out)
{
    __shared__ float s1[256], s2[256];
    int row = blockIdx.x, tid = threadIdx.x;
    float a0 = x[(size_t)row * CC + tid];
    float a1 = x[(size_t)row * CC + 256 + tid];
    s1[tid] = a0 + a1;
    s2[tid] = a0 * a0 + a1 * a1;
    __syncthreads();
    for (int s = 128; s > 0; s >>= 1) {
        if (tid < s) { s1[tid] += s1[tid + s]; s2[tid] += s2[tid + s]; }
        __syncthreads();
    }
    float mu   = s1[0] * (1.0f / CC);
    float var  = s2[0] * (1.0f / CC) - mu * mu;
    float rstd = rsqrtf(var + 1e-5f);
    out[(size_t)row * CC + tid]       = f2bf((a0 - mu) * rstd * g[tid]       + b[tid]);
    out[(size_t)row * CC + 256 + tid] = f2bf((a1 - mu) * rstd * g[tid + 256] + b[tid + 256]);
}

// ---------------------------------------------------------------------------
// Attention: per block = (b, h, 32-query tile); 2 waves x 16 queries each.
// scores (WMMA) -> mask -> softmax in LDS (unnormalized exp as bf16,
// 1/rowsum folded into the P@V accumulators) -> P@V (WMMA).
// ---------------------------------------------------------------------------
__global__ __launch_bounds__(64)
void attn_kernel(const __bf16* __restrict__ qb, const __bf16* __restrict__ kvb,
                 const float* __restrict__ mask, __bf16* __restrict__ outb)
{
    __shared__ float Pf[32][NS];       // 32 KB score scratch
    __shared__ float rowSum[32];

    const int tid = threadIdx.x, lane = tid & 31, wave = tid >> 5;
    const int blk = blockIdx.x;                 // b*8*128 + h*128 + ntile
    const int ntile = blk & 127;
    const int h = (blk >> 7) & 7;
    const int b = blk >> 10;
    const int nBase = ntile * 32 + wave * 16;
    const int r = lane & 15, half = lane >> 4;

    // --- q fragments: 16 queries x 64 (two K=32 chunks) ------------------
    v16bf qf[2];
#pragma unroll
    for (int c = 0; c < 2; ++c)
#pragma unroll
        for (int j = 0; j < 8; ++j) {
            int kd = c * 32 + ((j >= 4) ? 16 : 0) + half * 8 + (j & 3) * 2;
            const __bf16* p = qb + ((size_t)(b * NN + nBase + r)) * CC + h * DH + kd;
            unsigned u = *(const unsigned*)p;
            qf[c][2 * j]     = ((const __bf16*)&u)[0];
            qf[c][2 * j + 1] = ((const __bf16*)&u)[1];
        }

    // mask row base for this lane's first row (j = 0)
    const size_t mrow = (((size_t)(b * HEADS + h)) * NN + nBase + half * 8) * NS;

    // --- scores over 16 key tiles ---------------------------------------
    for (int t = 0; t < 16; ++t) {
        // prefetch next tile's mask line (dominant HBM stream)
        if (t < 15)
            __builtin_prefetch((const void*)(mask + mrow + (t + 1) * 16 + r), 0, 0);
        v8f acc;
#pragma unroll
        for (int j = 0; j < 8; ++j) acc[j] = 0.0f;
#pragma unroll
        for (int c = 0; c < 2; ++c) {
            v16bf kf;   // B = k^T : element (K=kd, col=m)
#pragma unroll
            for (int j = 0; j < 8; ++j) {
                int kd = c * 32 + ((j >= 4) ? 16 : 0) + half * 8 + (j & 3) * 2;
                const __bf16* p = kvb + ((size_t)(b * NS + t * 16 + r)) * 1024 + h * DH + kd;
                unsigned u = *(const unsigned*)p;
                kf[2 * j]     = ((const __bf16*)&u)[0];
                kf[2 * j + 1] = ((const __bf16*)&u)[1];
            }
            acc = __builtin_amdgcn_wmma_f32_16x16x32_bf16(
                false, qf[c], false, kf, (short)0, acc, false, false);
        }
#pragma unroll
        for (int j = 0; j < 8; ++j) {
            int n = nBase + half * 8 + j;
            int m = t * 16 + r;
            float mu = mask[(((size_t)(b * HEADS + h)) * NN + n) * NS + m];
            float s = acc[j] * SCALE + ((mu < 0.1f) ? -1e12f : 0.0f);
            Pf[wave * 16 + half * 8 + j][m] = s;
        }
    }
    __syncthreads();

    // --- softmax: one thread per row, write unnormalized exp as bf16 -----
    if (tid < 32) {
        float* rowp = Pf[tid];
        float mx = rowp[0];
        for (int i = 1; i < NS; ++i) mx = fmaxf(mx, rowp[i]);
        __bf16* rb = (__bf16*)rowp;            // packs into first half in-place
        float sum = 0.0f;
        for (int i = 0; i < NS; ++i) {
            float e = __expf(rowp[i] - mx);
            sum += e;
            rb[i] = f2bf(e);
        }
        rowSum[tid] = sum;
    }
    __syncthreads();

    // --- P @ V : K = 256 over 8 chunks, 4 output d-tiles -----------------
    v8f oacc[4];
#pragma unroll
    for (int nt = 0; nt < 4; ++nt)
#pragma unroll
        for (int j = 0; j < 8; ++j) oacc[nt][j] = 0.0f;

    for (int c = 0; c < 8; ++c) {
        v16bf pf;
#pragma unroll
        for (int j = 0; j < 8; ++j) {
            int kk = c * 32 + ((j >= 4) ? 16 : 0) + half * 8 + (j & 3) * 2;
            const __bf16* pp = ((const __bf16*)Pf[wave * 16 + r]) + kk;
            unsigned u = *(const unsigned*)pp;
            pf[2 * j]     = ((const __bf16*)&u)[0];
            pf[2 * j + 1] = ((const __bf16*)&u)[1];
        }
#pragma unroll
        for (int nt = 0; nt < 4; ++nt) {
            v16bf vf;   // B = v : element (K=m, col=dd)
#pragma unroll
            for (int j = 0; j < 8; ++j) {
                int m   = c * 32 + ((j >= 4) ? 16 : 0) + half * 8 + (j & 3) * 2;
                int col = nt * 16 + r;
                const __bf16* vp = kvb + ((size_t)(b * NS + m)) * 1024 + 512 + h * DH + col;
                vf[2 * j]     = vp[0];
                vf[2 * j + 1] = vp[1024];
            }
            oacc[nt] = __builtin_amdgcn_wmma_f32_16x16x32_bf16(
                false, pf, false, vf, (short)0, oacc[nt], false, false);
        }
    }

    // --- epilogue: fold 1/rowsum, store bf16 -----------------------------
#pragma unroll
    for (int nt = 0; nt < 4; ++nt)
#pragma unroll
        for (int j = 0; j < 8; ++j) {
            int n = nBase + half * 8 + j;
            float inv = 1.0f / rowSum[wave * 16 + half * 8 + j];
            outb[((size_t)(b * NN + n)) * CC + h * DH + nt * 16 + r] =
                f2bf(oacc[nt][j] * inv);
        }
}

// ---------------------------------------------------------------------------
// Host orchestration
// ---------------------------------------------------------------------------
extern "C" void kernel_launch(void* const* d_in, const int* in_sizes, int n_in,
                              void* d_out, int out_size, void* d_ws, size_t ws_size,
                              hipStream_t stream)
{
    const float* x      = (const float*)d_in[0];
    const float* x2     = (const float*)d_in[1];
    const float* mask_u = (const float*)d_in[2];
    const float* conv_w = (const float*)d_in[3];
    const float* conv_b = (const float*)d_in[4];
    const float* q_w    = (const float*)d_in[5];
    const float* q_b    = (const float*)d_in[6];
    const float* ln_g   = (const float*)d_in[7];
    const float* ln_b   = (const float*)d_in[8];
    const float* kv_w   = (const float*)d_in[9];
    const float* kv_b   = (const float*)d_in[10];
    const float* proj_w = (const float*)d_in[11];
    const float* proj_b = (const float*)d_in[12];

    char* ws = (char*)d_ws;
    size_t o = 0;
    auto alloc = [&](size_t bytes) { size_t r = o; o += (bytes + 255) & ~(size_t)255; return r; };

    const size_t BNC = (size_t)BB * NN * CC;                 // 16,777,216
    __bf16* xb   = (__bf16*)(ws + alloc(BNC * 2));            // x bf16 [B*N, C]
    __bf16* ac   = (__bf16*)(ws + alloc((size_t)BB * NS * KCONV * 2)); // im2row
    __bf16* cwb  = (__bf16*)(ws + alloc((size_t)KCONV * CC * 2));      // conv B
    __bf16* qwb  = (__bf16*)(ws + alloc((size_t)CC * CC * 2));
    __bf16* kwb  = (__bf16*)(ws + alloc((size_t)CC * 2 * CC * 2));
    __bf16* pwb  = (__bf16*)(ws + alloc((size_t)2 * CC * CC * 2));
    __bf16* qout = (__bf16*)(ws + alloc(BNC * 2));            // q [B*N, C]
    float*  kvx  = (float*) (ws + alloc((size_t)BB * NS * CC * 4));
    __bf16* kvn  = (__bf16*)(ws + alloc((size_t)BB * NS * CC * 2));
    __bf16* kvo  = (__bf16*)(ws + alloc((size_t)BB * NS * 2 * CC * 2)); // kv
    __bf16* aob  = (__bf16*)(ws + alloc(BNC * 2));            // attention out

    // ---- packing --------------------------------------------------------
    pack_bf16<<<(int)((BNC + 255) / 256), 256, 0, stream>>>(x, xb, (int)BNC);
    pack_bf16<<<(CC * CC + 255) / 256, 256, 0, stream>>>(q_w, qwb, CC * CC);
    pack_bf16<<<(CC * 2 * CC + 255) / 256, 256, 0, stream>>>(kv_w, kwb, CC * 2 * CC);
    pack_bf16<<<(2 * CC * CC + 255) / 256, 256, 0, stream>>>(proj_w, pwb, 2 * CC * CC);
    pack_convw<<<(KCONV * CC + 255) / 256, 256, 0, stream>>>(conv_w, cwb);
    pack_patches<<<(BB * NS * KCONV + 255) / 256, 256, 0, stream>>>(x2, ac);

    // ---- q = x @ q_w + q_b : [32768,512] x [512,512] -> bf16 ------------
    gemm_bf16<true><<<dim3(CC / 128, BB * NN / 128), 256, 0, stream>>>(
        xb, xb, CC, CC, CC, qwb, q_b, qout, BB * NN, CC, CC);

    // ---- conv-as-GEMM : [2048,8192] x [8192,512] -> fp32 ----------------
    gemm_bf16<false><<<dim3(CC / 128, BB * NS / 128), 256, 0, stream>>>(
        ac, ac, KCONV, KCONV, KCONV, cwb, conv_b, kvx, BB * NS, KCONV, CC);

    // ---- LayerNorm ------------------------------------------------------
    ln_kernel<<<BB * NS, 256, 0, stream>>>(kvx, ln_g, ln_b, kvn);

    // ---- kv = kvn @ kv_w + kv_b : [2048,512] x [512,1024] -> bf16 -------
    gemm_bf16<true><<<dim3(2 * CC / 128, BB * NS / 128), 256, 0, stream>>>(
        kvn, kvn, CC, CC, CC, kwb, kv_b, kvo, BB * NS, CC, 2 * CC);

    // ---- attention ------------------------------------------------------
    attn_kernel<<<BB * HEADS * (NN / 32), 64, 0, stream>>>(qout, kvo, mask_u, aob);

    // ---- final: concat([attn_out, x]) @ proj_w + proj_b -> fp32 d_out ---
    gemm_bf16<false><<<dim3(CC / 128, BB * NN / 128), 256, 0, stream>>>(
        aob, xb, CC, CC, CC, pwb, proj_b, (float*)d_out, BB * NN, 2 * CC, CC);
}
